// SetToSetPooling_80384607912066
// MI455X (gfx1250) — matrix operand order repaired and checked by reading, not compiled, not run on hardware
//
#include <hip/hip_runtime.h>
#include <hip/hip_bf16.h>

typedef __attribute__((ext_vector_type(16))) _Float16 v16h;
typedef __attribute__((ext_vector_type(8)))  float    v8f;

#define NG     2048      // graphs
#define NA     64        // atoms per graph
#define H      256       // feature dim
#define G4     1024      // 4*H gate width

// ---------------------------------------------------------------------------
// Prep: fuse W_ih + W_hh -> f16, packed directly in WMMA B-fragment layout.
// B_packed index = ((nt*8 + kc)*32 + lane)*16 + j
//   n (gate col) = nt*16 + (lane & 15)
//   k            = kc*32 + (lane>>4)*16 + j     (contiguous k order, same as the A fragment)
// Also fuse biases: bsum[n] = b_ih[n] + b_hh[n].
// ---------------------------------------------------------------------------
__global__ __launch_bounds__(256) void s2s_prep_kernel(
    const float* __restrict__ W_ih, const float* __restrict__ W_hh,
    const float* __restrict__ b_ih, const float* __restrict__ b_hh,
    _Float16* __restrict__ Bp, float* __restrict__ bsum)
{
    const int id = blockIdx.x * 256 + threadIdx.x;   // 64*8*32*16 = 262144
    if (id < G4) bsum[id] = b_ih[id] + b_hh[id];
    const int j    = id & 15;
    const int lane = (id >> 4) & 31;
    const int kc   = (id >> 9) & 7;
    const int nt   = id >> 12;
    const int n    = nt * 16 + (lane & 15);
    const int k    = kc * 32 + ((lane >> 4) * 16) + j;
    Bp[id] = (_Float16)(W_ih[n * H + k] + W_hh[n * H + k]);
}

__global__ __launch_bounds__(256) void s2s_init_kernel(
    float* __restrict__ h, float* __restrict__ c)
{
    const int id = blockIdx.x * 256 + threadIdx.x;   // NG*H
    h[id] = 0.0f;
    c[id] = 0.0f;
}

// ---------------------------------------------------------------------------
// LSTM step: gates = h @ Wsum^T + bsum; q = sig(o)*tanh(c'); c' = sig(f)*c + sig(i)*tanh(g)
// One WG per 16 graphs. 8 waves; wave w handles column groups jj = 2w, 2w+1,
// and for each jj owns the i/f/g/o tiles (nt = jj, jj+16, jj+32, jj+48) so the
// gate nonlinearity is wave-local on the WMMA accumulators.
// ---------------------------------------------------------------------------
__global__ __launch_bounds__(256) void s2s_lstm_kernel(
    const float* __restrict__ h, float* __restrict__ c, float* __restrict__ q,
    const v16h* __restrict__ Bp, const float* __restrict__ bsum)
{
    __shared__ _Float16 hs[16][264];   // 16x256 f16 tile of h, padded rows
    const int tile = blockIdx.x;       // 0..127 -> graphs tile*16 ..
    const int tid  = threadIdx.x;

    // Stage h tile to LDS as f16 (coalesced)
#pragma unroll
    for (int i = 0; i < 16; ++i)
        hs[i][tid] = (_Float16)h[(size_t)(tile * 16 + i) * H + tid];
    __syncthreads();

    const int w    = tid >> 5;
    const int lane = tid & 31;
    const int m    = lane & 15;   // A-matrix row / D-matrix column
    const int g    = lane >> 4;   // lane half

    union AFrag { v16h v; unsigned u[8]; };

#pragma unroll
    for (int s = 0; s < 2; ++s) {
        const int jj = w * 2 + s;                 // 0..15 column group
        v8f ai = {}, af = {}, ag = {}, ao = {};

#pragma unroll
        for (int kc = 0; kc < 8; ++kc) {
            // A fragment (16x32 f16): K = kc*32 + g*16 + j
            AFrag A;
            const _Float16* hrow = &hs[m][0];
            const int o1 = kc * 32 + g * 16;
#pragma unroll
            for (int p = 0; p < 4; ++p) {
                A.u[p]     = *(const unsigned*)(hrow + o1 + 2 * p);
                A.u[4 + p] = *(const unsigned*)(hrow + o1 + 8 + 2 * p);
            }
            // B fragments for the four gates (pre-packed, 32B/lane, coalesced)
            const v16h bi = Bp[((jj     ) * 8 + kc) * 32 + lane];
            const v16h bf = Bp[((jj + 16) * 8 + kc) * 32 + lane];
            const v16h bg = Bp[((jj + 32) * 8 + kc) * 32 + lane];
            const v16h bo = Bp[((jj + 48) * 8 + kc) * 32 + lane];

            ai = __builtin_amdgcn_wmma_f32_16x16x32_f16(false, A.v, false, bi, (short)0, ai, false, false);
            af = __builtin_amdgcn_wmma_f32_16x16x32_f16(false, A.v, false, bf, (short)0, af, false, false);
            ag = __builtin_amdgcn_wmma_f32_16x16x32_f16(false, A.v, false, bg, (short)0, ag, false, false);
            ao = __builtin_amdgcn_wmma_f32_16x16x32_f16(false, A.v, false, bo, (short)0, ao, false, false);
        }

        // Gate nonlinearities. D layout: vgpr r, lanes 0-15 -> M=r, lanes 16-31 -> M=8+r, N=lane&15.
        const int col = jj * 16 + m;              // 0..255 within each gate block
        const float bi0 = bsum[col];
        const float bf0 = bsum[H + col];
        const float bg0 = bsum[2 * H + col];
        const float bo0 = bsum[3 * H + col];
#pragma unroll
        for (int r = 0; r < 8; ++r) {
            const int mm  = r + (g ? 8 : 0);
            const size_t gid = (size_t)(tile * 16 + mm) * H + col;
            const float iv = ai[r] + bi0;
            const float fv = af[r] + bf0;
            const float gv = ag[r] + bg0;
            const float ov = ao[r] + bo0;
            const float si = 1.0f / (1.0f + __expf(-iv));
            const float sf = 1.0f / (1.0f + __expf(-fv));
            const float so = 1.0f / (1.0f + __expf(-ov));
            const float cn = sf * c[gid] + si * tanhf(gv);
            c[gid] = cn;
            q[gid] = so * tanhf(cn);
        }
    }
}

// ---------------------------------------------------------------------------
// Attention: scores = x . q[graph]; segment softmax over 64 atoms; r = attn^T x.
// One WG (8 waves) per graph. x is L2-resident (134MB < 192MB L2).
// ---------------------------------------------------------------------------
__global__ __launch_bounds__(256) void s2s_attn_kernel(
    const float* __restrict__ x, const float* __restrict__ q,
    float* __restrict__ rout)
{
    __shared__ float qsh[H];
    __shared__ float sc[NA];
    const int b   = blockIdx.x;
    const int tid = threadIdx.x;

    qsh[tid] = q[(size_t)b * H + tid];
    __syncthreads();

    const int w    = tid >> 5;
    const int lane = tid & 31;
    const float* xg = x + (size_t)b * NA * H;

    // Phase 1: wave-per-atom coalesced dot products (8 atoms per wave)
    for (int ia = 0; ia < 8; ++ia) {
        const int a = w * 8 + ia;
        const float* xa = xg + a * H;
        float p = 0.0f;
#pragma unroll
        for (int i = 0; i < 8; ++i) {
            const int k = lane + 32 * i;
            p += xa[k] * qsh[k];
        }
#pragma unroll
        for (int off = 16; off; off >>= 1) p += __shfl_xor(p, off, 32);
        if (lane == 0) sc[a] = p;
    }
    __syncthreads();

    // Phase 2: softmax over 64 scores (one wave, butterfly reductions)
    if (tid < 32) {
        const float s1 = sc[tid], s2 = sc[tid + 32];
        float mx = fmaxf(s1, s2);
#pragma unroll
        for (int off = 16; off; off >>= 1) mx = fmaxf(mx, __shfl_xor(mx, off, 32));
        const float e1 = __expf(s1 - mx), e2 = __expf(s2 - mx);
        float ss = e1 + e2;
#pragma unroll
        for (int off = 16; off; off >>= 1) ss += __shfl_xor(ss, off, 32);
        const float inv = 1.0f / ss;
        sc[tid]      = e1 * inv;
        sc[tid + 32] = e2 * inv;
    }
    __syncthreads();

    // Phase 3: weighted sum over atoms (coalesced across feature dim)
    float r = 0.0f;
#pragma unroll 8
    for (int a = 0; a < NA; ++a) r += xg[a * H + tid] * sc[a];
    rout[(size_t)b * H + tid] = r;
}

// Output: concat([q, r], -1) -> [NG, 2H] fp32
__global__ __launch_bounds__(256) void s2s_out_kernel(
    const float* __restrict__ q, const float* __restrict__ r,
    float* __restrict__ out)
{
    const int id = blockIdx.x * 256 + threadIdx.x;   // NG*512
    const int b = id >> 9, j = id & 511;
    out[id] = (j < H) ? q[(size_t)b * H + j] : r[(size_t)b * H + (j - H)];
}

extern "C" void kernel_launch(void* const* d_in, const int* in_sizes, int n_in,
                              void* d_out, int out_size, void* d_ws, size_t ws_size,
                              hipStream_t stream) {
    const float* x    = (const float*)d_in[0];
    // d_in[1]=batch, d_in[2]=sizes: structure is uniform (64 atoms/graph) -> unused
    const float* W_ih = (const float*)d_in[3];
    const float* W_hh = (const float*)d_in[4];
    const float* b_ih = (const float*)d_in[5];
    const float* b_hh = (const float*)d_in[6];
    float* out = (float*)d_out;

    // Workspace layout (floats): h | c | q | bsum | B_packed(f16)
    float* ws   = (float*)d_ws;
    float* h    = ws;
    float* c    = ws + (size_t)NG * H;
    float* q    = ws + (size_t)2 * NG * H;
    float* bsum = ws + (size_t)3 * NG * H;
    _Float16* Bp = (_Float16*)(bsum + G4);   // 32B-aligned (offset 6295552 bytes)

    // Fused weights/bias into WMMA-ready layout
    s2s_prep_kernel<<<(G4 / 16) * (H / 32) * 32 * 16 / 256, 256, 0, stream>>>(
        W_ih, W_hh, b_ih, b_hh, Bp, bsum);
    // Zero h, c (required every call: ws is not re-poisoned)
    s2s_init_kernel<<<(NG * H) / 256, 256, 0, stream>>>(h, c);

    for (int step = 0; step < 3; ++step) {
        s2s_lstm_kernel<<<NG / 16, 256, 0, stream>>>(h, c, q, (const v16h*)Bp, bsum);
        s2s_attn_kernel<<<NG, 256, 0, stream>>>(x, q, h);   // h <- r
    }

    s2s_out_kernel<<<(NG * 2 * H) / 256, 256, 0, stream>>>(q, h, out);
}
